// NaiveV2E_10290741641948
// MI455X (gfx1250) — matrix operand order repaired and checked by reading, not compiled, not run on hardware
//
#include <hip/hip_runtime.h>

// Problem constants (match reference)
#define N_NODES 8192
#define N_EDGES 8192
#define DIM     128
#define NORD    17   // MAX_L + 1

typedef __attribute__((ext_vector_type(16))) _Float16 v16h;
typedef __attribute__((ext_vector_type(8)))  _Float16 v8h;
typedef __attribute__((ext_vector_type(8)))  float    v8f;
typedef __attribute__((ext_vector_type(4)))  float    v4f;

// Build a 16-half fragment from two contiguous 16-byte runs (LDS or global).
__device__ inline v16h frag16(const _Float16* p0, const _Float16* p1) {
  v8h lo = *(const v8h*)p0;
  v8h hi = *(const v8h*)p1;
  return __builtin_shufflevector(lo, hi, 0,1,2,3,4,5,6,7,8,9,10,11,12,13,14,15);
}

// ---------------------------------------------------------------- x0 = sum(x, axis=0)
__global__ void zero_x0_kernel(float* __restrict__ x0) {
  x0[threadIdx.x] = 0.0f;
}

__global__ void colsum_kernel(const float* __restrict__ x, float* __restrict__ x0) {
  // 64 blocks x 256 threads; block handles 128 rows, threads stride columns.
  int t   = threadIdx.x;
  int col = t & (DIM - 1);
  int r0  = t >> 7;                    // 0 or 1
  long base = (long)blockIdx.x * 128 * DIM;
  float acc = 0.0f;
  for (int r = r0; r < 128; r += 2)
    acc += x[base + (long)r * DIM + col];
  atomicAdd(&x0[col], acc);
}

// ------------------------------------------- y0[l][j] = sum_i (x0[i]/N) * W[0][l][i][j]
__global__ void y0_kernel(const float* __restrict__ x0, const float* __restrict__ W,
                          float* __restrict__ y0) {
  __shared__ float xs[DIM];
  int j = threadIdx.x;
  xs[j] = x0[j] * (1.0f / (float)N_NODES);
  __syncthreads();
  int l = blockIdx.x;
  const float* Wl = W + (long)l * DIM * DIM;   // W[0][l]
  float acc = 0.0f;
#pragma unroll 8
  for (int i = 0; i < DIM; ++i)
    acc = fmaf(xs[i], Wl[i * DIM + j], acc);
  y0[l * DIM + j] = acc;
}

// --------------------------- xT_h[d][n] = (f16) x[n][d]   (tiled 64x64 transpose)
__global__ void xT_kernel(const float* __restrict__ x, _Float16* __restrict__ xt) {
  __shared__ __attribute__((aligned(16))) _Float16 tile[64][64 + 8];
  int t = threadIdx.x;                       // 256
  long n0 = (long)blockIdx.x * 64;
  int  d0 = blockIdx.y * 64;
#pragma unroll
  for (int p = 0; p < 16; ++p) {
    int idx = p * 256 + t;
    int dd = idx & 63, nn = idx >> 6;
    tile[dd][nn] = (_Float16)x[(n0 + nn) * DIM + d0 + dd];
  }
  __syncthreads();
#pragma unroll
  for (int p = 0; p < 2; ++p) {
    int c  = p * 256 + t;                    // 512 chunks of 8 halfs
    int dd = c >> 3, nc = (c & 7) * 8;
    v8h v = *(const v8h*)&tile[dd][nc];
    *(v8h*)&xt[(long)(d0 + dd) * N_NODES + n0 + nc] = v;
  }
}

// ---------------------------- out_v = x @ W[1][1] + y0[1] + B[1]   (WMMA f16 -> f32)
__global__ void node_gemm_kernel(const float* __restrict__ x, const float* __restrict__ W,
                                 const float* __restrict__ Bb, const float* __restrict__ y0,
                                 float* __restrict__ out_v) {
  __shared__ __attribute__((aligned(16))) _Float16 WsT[DIM][DIM + 8];  // [n][k]
  __shared__ __attribute__((aligned(16))) _Float16 AxsT[16][DIM + 8];  // [m][k]
  const float* W11 = W + (long)(NORD + 1) * DIM * DIM;   // W[1][1]
  int t = threadIdx.x;
#pragma unroll
  for (int p = 0; p < 64; ++p) {
    int idx = p * 256 + t;
    int k = idx >> 7, n = idx & 127;
    WsT[n][k] = (_Float16)W11[idx];
  }
  long m0 = (long)blockIdx.x * 16;
#pragma unroll
  for (int p = 0; p < 8; ++p) {
    int idx = p * 256 + t;
    int r = idx >> 7, c = idx & 127;
    AxsT[r][c] = (_Float16)x[(m0 + r) * DIM + c];
  }
  __syncthreads();

  int lane = t & 31, wave = t >> 5;
  int m = lane & 15, h = lane >> 4;
  int n0 = wave * 16;                  // 8 waves cover all 128 output dims
  v8f acc = {};
#pragma unroll
  for (int k0 = 0; k0 < DIM; k0 += 32) {
    v16h a = frag16(&AxsT[m][k0 + 16 * h],     &AxsT[m][k0 + 16 * h + 8]);
    v16h b = frag16(&WsT[n0 + m][k0 + 16 * h], &WsT[n0 + m][k0 + 16 * h + 8]);
    acc = __builtin_amdgcn_wmma_f32_16x16x32_f16(false, a, false, b, (short)0, acc,
                                                 false, false);
  }
  float addn = y0[DIM + n0 + m] + Bb[DIM + n0 + m];   // y0[1][n] + B[1][n]
#pragma unroll
  for (int r = 0; r < 8; ++r) {
    int M = 8 * h + r;
    out_v[(m0 + M) * DIM + n0 + m] = acc[r] + addn;
  }
}

// ---------------------------------- x1 = incidence^T @ x  (E x D, raw sums, WMMA f16)
// Software-pipelined: one NT global_load_b128 of the next A-slab per thread overlaps
// the WMMAs on the current slab; LDS double-buffered -> single barrier per slab.
// B fragments come directly from the precomputed f16 xT (L2-resident, 32B/lane).
__global__ void edge_gemm_kernel(const float* __restrict__ inc,
                                 const _Float16* __restrict__ xt,
                                 float* __restrict__ x1) {
  __shared__ __attribute__((aligned(16))) _Float16 AsT[2][16][64 + 8]; // [buf][edge][k]
  int t = threadIdx.x;
  long e0 = (long)blockIdx.x * 16;
  int lane = t & 31, wave = t >> 5;
  int m = lane & 15, h = lane >> 4;
  int n0 = wave * 16;
  const _Float16* xrow = xt + (long)(n0 + m) * N_NODES;   // this lane's B row

  // Stager mapping: thread owns slab-row k = t>>2, edges e = (t&3)*4 .. +3.
  int kl = t >> 2;              // 0..63
  int el = (t & 3) << 2;        // 0,4,8,12
  const float* aptr = inc + (long)kl * N_EDGES + e0 + el;
  v4f pre = __builtin_nontemporal_load((const v4f*)aptr);

  v8f acc = {};
  int buf = 0;
  for (int k0 = 0; k0 < N_NODES; k0 += 64, buf ^= 1) {
    // Commit the prefetched slab to LDS (transposed, f16).
    AsT[buf][el + 0][kl] = (_Float16)pre[0];
    AsT[buf][el + 1][kl] = (_Float16)pre[1];
    AsT[buf][el + 2][kl] = (_Float16)pre[2];
    AsT[buf][el + 3][kl] = (_Float16)pre[3];
    // Kick off the next slab's load; it completes under the WMMAs below.
    if (k0 + 64 < N_NODES) {
      aptr += (long)64 * N_EDGES;
      pre = __builtin_nontemporal_load((const v4f*)aptr);
      if ((t & 3) == 0 && k0 + 128 < N_NODES)
        __builtin_prefetch(aptr + (long)64 * N_EDGES, 0, 0);
    }
    __syncthreads();
#pragma unroll
    for (int kk = 0; kk < 64; kk += 32) {
      v16h a = frag16(&AsT[buf][m][kk + 16 * h], &AsT[buf][m][kk + 16 * h + 8]);
      const _Float16* bp = xrow + k0 + kk + 16 * h;
      v16h b = frag16(bp, bp + 8);
      acc = __builtin_amdgcn_wmma_f32_16x16x32_f16(false, a, false, b, (short)0, acc,
                                                   false, false);
    }
  }
#pragma unroll
  for (int r = 0; r < 8; ++r) {
    int M = 8 * h + r;
    x1[(e0 + M) * DIM + n0 + m] = acc[r];
  }
}

// ---- out_e[e] = y0[ord_e] + (x1[e]/norm_e) @ W[1][ord_e] + B[ord_e]  (VALU, gathered W)
__global__ void edge_final_kernel(const float* __restrict__ x1, const int* __restrict__ ord,
                                  const float* __restrict__ norm, const float* __restrict__ W,
                                  const float* __restrict__ Bb, const float* __restrict__ y0,
                                  float* __restrict__ out_e) {
  __shared__ float xe[DIM];
  int e = blockIdx.x, j = threadIdx.x;
  int l = ord[e];
  float inv = 1.0f / norm[e];
  xe[j] = x1[(long)e * DIM + j] * inv;
  __syncthreads();
  const float* W1l = W + (long)(NORD + l) * DIM * DIM;   // W[1][l]
  float acc = 0.0f;
#pragma unroll 8
  for (int i = 0; i < DIM; ++i)
    acc = fmaf(xe[i], W1l[i * DIM + j], acc);            // coalesced over j
  out_e[(long)e * DIM + j] = y0[l * DIM + j] + acc + Bb[l * DIM + j];
}

extern "C" void kernel_launch(void* const* d_in, const int* in_sizes, int n_in,
                              void* d_out, int out_size, void* d_ws, size_t ws_size,
                              hipStream_t stream) {
  const float* x    = (const float*)d_in[0];  // (N, D)
  const float* inc  = (const float*)d_in[1];  // (N, E)
  const int*   ord  = (const int*)d_in[2];    // (E,)
  const float* norm = (const float*)d_in[3];  // (E,)
  const float* W    = (const float*)d_in[4];  // (2, 17, D, D)
  const float* Bb   = (const float*)d_in[5];  // (17, D)

  float* out_v = (float*)d_out;                         // (N, D)
  float* out_e = (float*)d_out + (long)N_NODES * DIM;   // (E, D)

  float*     x0 = (float*)d_ws;                 // 128
  float*     y0 = x0 + DIM;                     // 17 * 128
  float*     x1 = y0 + NORD * DIM;              // E * D raw sums
  _Float16*  xt = (_Float16*)(x1 + (long)N_EDGES * DIM);  // D x N f16 transpose

  zero_x0_kernel<<<1, DIM, 0, stream>>>(x0);
  colsum_kernel<<<64, 256, 0, stream>>>(x, x0);
  y0_kernel<<<NORD, DIM, 0, stream>>>(x0, W, y0);
  xT_kernel<<<dim3(N_NODES / 64, DIM / 64), 256, 0, stream>>>(x, xt);
  node_gemm_kernel<<<N_NODES / 16, 256, 0, stream>>>(x, W, Bb, y0, out_v);
  edge_gemm_kernel<<<N_EDGES / 16, 256, 0, stream>>>(inc, xt, x1);
  edge_final_kernel<<<N_EDGES, DIM, 0, stream>>>(x1, ord, norm, W, Bb, y0, out_e);
}
